// Swin3D_60533269069902
// MI455X (gfx1250) — matrix operand order repaired and checked by reading, not compiled, not run on hardware
//
#include <hip/hip_runtime.h>

typedef __bf16 bf16;
typedef float  v8f   __attribute__((ext_vector_type(8)));
typedef bf16   v16bf __attribute__((ext_vector_type(16)));
typedef bf16   v8bf  __attribute__((ext_vector_type(8)));
typedef unsigned int u32x4 __attribute__((ext_vector_type(4)));
typedef int          i32x4 __attribute__((ext_vector_type(4)));
typedef int          i32x8 __attribute__((ext_vector_type(8)));

#define NN 32768
#define KN 16
#define DD 128

__device__ __forceinline__ v8f wmma_bf16(v16bf a, v16bf b, v8f c) {
  return __builtin_amdgcn_wmma_f32_16x16x32_bf16(false, a, false, b, (short)0, c, false, false);
}
__device__ __forceinline__ v16bf cat8(v8bf lo, v8bf hi) {
  return __builtin_shufflevector(lo, hi, 0,1,2,3,4,5,6,7,8,9,10,11,12,13,14,15);
}
__device__ __forceinline__ v16bf ldb(const bf16* p) {
  return cat8(*(const v8bf*)p, *(const v8bf*)(p + 16));
}
__device__ __forceinline__ void st4(float* d, float4 v){ d[0]=v.x; d[1]=v.y; d[2]=v.z; d[3]=v.w; }
__device__ __forceinline__ float4 ld4(const float* p){ return *(const float4*)p; }
__device__ __forceinline__ v8f zero8(){
  v8f z;
#pragma unroll
  for (int i = 0; i < 8; i++) z[i] = 0.f;
  return z;
}

// ---- Tensor Data Mover: 1-row 2D descriptor, global -> LDS, ndwords <= 65535 ----
__device__ __forceinline__ void tdm_load_to_lds(unsigned lds_off, const void* gaddr,
                                                unsigned ndwords) {
  unsigned long long ga = (unsigned long long)(uintptr_t)gaddr;
  u32x4 g0;
  g0[0] = 1u;                                   // count=1, user descriptor
  g0[1] = lds_off;                              // lds_addr
  g0[2] = (unsigned)(ga & 0xffffffffu);         // global_addr[31:0]
  g0[3] = (unsigned)((ga >> 32) & 0x01ffffffu)  // global_addr[56:32]
        | (2u << 30);                           // type = 2 ("image")
  i32x8 g1;
  g1[0] = (2 << 16);                            // wg_mask=0, data_size=4B
  g1[1] = (int)((ndwords & 0xffffu) << 16);     // tensor_dim0[15:0]
  g1[2] = (int)(((ndwords >> 16) & 0xffffu) | (1u << 16)); // tensor_dim0 hi, tensor_dim1=1
  g1[3] = (int)((ndwords & 0xffffu) << 16);     // tile_dim0 = ndwords
  g1[4] = 1;                                    // tile_dim1 = 1
  g1[5] = (int)ndwords;                         // tensor_dim0_stride lo
  g1[6] = 0;
  g1[7] = 0;
  i32x4 gz = {0, 0, 0, 0};
#if defined(__clang_major__) && (__clang_major__ >= 23)
  i32x8 gz8 = {0, 0, 0, 0, 0, 0, 0, 0};
  __builtin_amdgcn_tensor_load_to_lds(g0, g1, gz, gz, gz8, 0);
#else
  __builtin_amdgcn_tensor_load_to_lds(g0, g1, gz, gz, 0);
#endif
}

// ---------------- prep ----------------
__global__ void k_zero(float* p, int n) {
  int i = blockIdx.x * blockDim.x + threadIdx.x;
  if (i < n) p[i] = 0.f;
}

__global__ void k_transpose_bf16(const float* __restrict__ in, bf16* __restrict__ out,
                                 int R, int C) {
  int i = blockIdx.x * blockDim.x + threadIdx.x;
  if (i < R * C) {
    int r = i / C, c = i % C;
    out[c * R + r] = (bf16)in[r * C + c];
  }
}

// ---------------- edge MLP: one wave per node ----------------
__global__ __launch_bounds__(256) void k_edge(
    const float* __restrict__ h, const int* __restrict__ nbr,
    const bf16* __restrict__ W1t, const float* __restrict__ b1,
    const float* __restrict__ W2, const float* __restrict__ b2,
    float* __restrict__ stats)
{
  __shared__ bf16 sW1[DD * DD];  // 32 KB, W1 transposed (row = out feature)
  // Stage W1t into LDS with the Tensor Data Mover (one DMA, issued by wave 0).
  if (threadIdx.x < 32) {
    tdm_load_to_lds((unsigned)(uintptr_t)(void*)sW1, W1t, DD * DD / 2);
    __builtin_amdgcn_s_wait_tensorcnt((short)0);
  }
  __syncthreads();

  const int wave = threadIdx.x >> 5;
  const int lane = threadIdx.x & 31;
  const int m    = lane & 15;   // edge row (A matrix row)
  const int hi   = lane >> 4;   // half select
  const int node = blockIdx.x * 8 + wave;

  const int sidx = nbr[node * KN + m];
  const float* srow = h + (size_t)sidx * DD;
  const float* drow = h + (size_t)node * DD;

  v8f acc[8];
#pragma unroll
  for (int j = 0; j < 8; j++) acc[j] = zero8();

#pragma unroll
  for (int c = 0; c < 4; c++) {
    const int f0 = c * 32 + hi * 8;
    float sv[16], dv[16];
    st4(sv + 0,  ld4(srow + f0));      st4(sv + 4,  ld4(srow + f0 + 4));
    st4(sv + 8,  ld4(srow + f0 + 16)); st4(sv + 12, ld4(srow + f0 + 20));
    st4(dv + 0,  ld4(drow + f0));      st4(dv + 4,  ld4(drow + f0 + 4));
    st4(dv + 8,  ld4(drow + f0 + 16)); st4(dv + 12, ld4(drow + f0 + 20));
    v16bf a;
#pragma unroll
    for (int i = 0; i < 16; i++) a[i] = (bf16)(sv[i] - dv[i]);
#pragma unroll
    for (int j = 0; j < 8; j++) {
      const bf16* bp = sW1 + (16 * j + m) * DD + c * 32 + hi * 8;
      acc[j] = wmma_bf16(a, ldb(bp), acc[j]);
    }
  }

  // att[m] = relu( relu(a) . W2 + b2 ); w = exp(-att); reduce stats over 16 edges
  float b1v[8], w2v[8];
#pragma unroll
  for (int j = 0; j < 8; j++) { b1v[j] = b1[16 * j + m]; w2v[j] = W2[16 * j + m]; }
  const float b2s = b2[0];

  float sumw = 0.f, maxw = -3.4e38f, minw = 3.4e38f;
#pragma unroll
  for (int r = 0; r < 8; r++) {
    float t = 0.f;
#pragma unroll
    for (int j = 0; j < 8; j++) {
      float av = acc[j][r] + b1v[j];
      av = av > 0.f ? av : 0.f;
      t += av * w2v[j];
    }
    t += __shfl_xor(t, 1); t += __shfl_xor(t, 2);
    t += __shfl_xor(t, 4); t += __shfl_xor(t, 8);
    t += b2s; t = t > 0.f ? t : 0.f;
    float w = __expf(-t);
    sumw += w; maxw = fmaxf(maxw, w); minw = fminf(minw, w);
  }
  sumw += __shfl_xor(sumw, 16);
  maxw = fmaxf(maxw, __shfl_xor(maxw, 16));
  minw = fminf(minw, __shfl_xor(minw, 16));

  if (lane == 0) {
    stats[node * 3 + 0] = sumw * (1.f / 16.f);
    stats[node * 3 + 1] = maxw;
    stats[node * 3 + 2] = minw;
  }
}

// ---------------- out = [mean_agg|max_agg] @ OW + Ob ; tmp = h + out ; BN1 sums ----------------
__global__ __launch_bounds__(256) void k_out(
    const float* __restrict__ h, const float* __restrict__ stats,
    const bf16* __restrict__ OWt, const float* __restrict__ Ob,
    float* __restrict__ tmp, float* __restrict__ colsum)
{
  __shared__ float sAcc[256];
  for (int i = threadIdx.x; i < 256; i += 256) sAcc[i] = 0.f;
  __syncthreads();

  const int wave = threadIdx.x >> 5, lane = threadIdx.x & 31;
  const int m = lane & 15, hi = lane >> 4;
  const int tile = blockIdx.x * 8 + wave;
  const int node = tile * 16 + m;
  const float* hrow = h + (size_t)node * DD;
  const float meanw = stats[node * 3 + 0];
  const float maxw  = stats[node * 3 + 1];
  const float minw  = stats[node * 3 + 2];

  v8f acc[8];
#pragma unroll
  for (int j = 0; j < 8; j++) acc[j] = zero8();

#pragma unroll
  for (int c = 0; c < 8; c++) {
    const int f0 = c * 32 + hi * 8;          // concat feature in [0,256)
    const int g0 = (c < 4) ? f0 : f0 - 128;  // feature in h
    float hv[16];
    st4(hv + 0,  ld4(hrow + g0));      st4(hv + 4,  ld4(hrow + g0 + 4));
    st4(hv + 8,  ld4(hrow + g0 + 16)); st4(hv + 12, ld4(hrow + g0 + 20));
    v16bf a;
    if (c < 4) {
#pragma unroll
      for (int i = 0; i < 16; i++) a[i] = (bf16)(meanw * hv[i]);
    } else {
#pragma unroll
      for (int i = 0; i < 16; i++) a[i] = (bf16)(hv[i] * (hv[i] >= 0.f ? maxw : minw));
    }
#pragma unroll
    for (int j = 0; j < 8; j++) {
      const bf16* bp = OWt + (16 * j + m) * 256 + c * 32 + hi * 8;
      acc[j] = wmma_bf16(a, ldb(bp), acc[j]);
    }
  }

  const int rowbase = tile * 16 + hi * 8;
#pragma unroll
  for (int j = 0; j < 8; j++) {
    const int f = 16 * j + m;
    const float ob = Ob[f];
    float s = 0.f, q = 0.f;
#pragma unroll
    for (int r = 0; r < 8; r++) {
      const int nd = rowbase + r;
      const float t = h[(size_t)nd * DD + f] + acc[j][r] + ob;
      tmp[(size_t)nd * DD + f] = t;
      s += t; q += t * t;
    }
    s += __shfl_xor(s, 16); q += __shfl_xor(q, 16);
    if (hi == 0) { atomicAdd(&sAcc[f], s); atomicAdd(&sAcc[128 + f], q); }
  }
  __syncthreads();
  for (int i = threadIdx.x; i < 256; i += 256) atomicAdd(&colsum[i], sAcc[i]);
}

// ---------------- fold BN stats into scale/shift ----------------
__global__ void k_bnscale(const float* __restrict__ colsum,
                          const float* __restrict__ gamma, const float* __restrict__ beta,
                          float* __restrict__ scale, float* __restrict__ shift)
{
  const int f = threadIdx.x;
  if (f < DD) {
    const float mu  = colsum[f] * (1.f / NN);
    const float var = colsum[128 + f] * (1.f / NN) - mu * mu;
    const float inv = rsqrtf(var + 1e-5f);
    const float sc  = gamma[f] * inv;
    scale[f] = sc;
    shift[f] = beta[f] - mu * sc;
  }
}

// ---------------- fused FFN: x=BN1(tmp); y=relu(x@F1W+b)@F2W+b; out=x+y; BN2 sums ----------------
__global__ __launch_bounds__(256) void k_ffn(
    const float* __restrict__ tmp,
    const float* __restrict__ scale1, const float* __restrict__ shift1,
    const bf16* __restrict__ F1Wt, const float* __restrict__ F1b,
    const bf16* __restrict__ F2Wt, const float* __restrict__ F2b,
    float* __restrict__ out, float* __restrict__ colsum2)
{
  __shared__ bf16 sU[8 * 16 * 256];  // 64 KB: per-wave 16x256 activation tile
  __shared__ float sAcc[256];
  for (int i = threadIdx.x; i < 256; i += 256) sAcc[i] = 0.f;
  __syncthreads();

  const int wave = threadIdx.x >> 5, lane = threadIdx.x & 31;
  const int m = lane & 15, hi = lane >> 4;
  const int tile = blockIdx.x * 8 + wave;
  const int node = tile * 16 + m;
  const float* trow = tmp + (size_t)node * DD;
  bf16* uRow = sU + wave * 16 * 256;

  // A = x (bf16), built from tmp with BN1 folded in
  v16bf Ax[4];
#pragma unroll
  for (int c = 0; c < 4; c++) {
    const int f0 = c * 32 + hi * 8;
    float tv[16], sc[16], sh[16];
    st4(tv + 0,  ld4(trow + f0));        st4(tv + 4,  ld4(trow + f0 + 4));
    st4(tv + 8,  ld4(trow + f0 + 16));   st4(tv + 12, ld4(trow + f0 + 20));
    st4(sc + 0,  ld4(scale1 + f0));      st4(sc + 4,  ld4(scale1 + f0 + 4));
    st4(sc + 8,  ld4(scale1 + f0 + 16)); st4(sc + 12, ld4(scale1 + f0 + 20));
    st4(sh + 0,  ld4(shift1 + f0));      st4(sh + 4,  ld4(shift1 + f0 + 4));
    st4(sh + 8,  ld4(shift1 + f0 + 16)); st4(sh + 12, ld4(shift1 + f0 + 20));
#pragma unroll
    for (int i = 0; i < 16; i++) Ax[c][i] = (bf16)(sc[i] * tv[i] + sh[i]);
  }

  // GEMM1: u = relu(x @ F1W + F1b), N=256 done in two halves of 128
#pragma unroll
  for (int nh = 0; nh < 2; nh++) {
    v8f acc[8];
#pragma unroll
    for (int j = 0; j < 8; j++) acc[j] = zero8();
#pragma unroll
    for (int c = 0; c < 4; c++) {
#pragma unroll
      for (int j = 0; j < 8; j++) {
        const int f = nh * 128 + 16 * j + m;
        const bf16* bp = F1Wt + (size_t)f * DD + c * 32 + hi * 8;
        acc[j] = wmma_bf16(Ax[c], ldb(bp), acc[j]);
      }
    }
#pragma unroll
    for (int j = 0; j < 8; j++) {
      const int f = nh * 128 + 16 * j + m;
      const float bb = F1b[f];
#pragma unroll
      for (int r = 0; r < 8; r++) {
        float v = acc[j][r] + bb;
        v = v > 0.f ? v : 0.f;
        uRow[(r + hi * 8) * 256 + f] = (bf16)v;
      }
    }
  }
  __syncthreads();

  // GEMM2: y = u @ F2W
  v8f acc2[8];
#pragma unroll
  for (int j = 0; j < 8; j++) acc2[j] = zero8();
#pragma unroll
  for (int c = 0; c < 8; c++) {
    const int f0 = c * 32 + hi * 8;
    const bf16* ap = uRow + m * 256 + f0;
    v16bf a = cat8(*(const v8bf*)ap, *(const v8bf*)(ap + 16));
#pragma unroll
    for (int j = 0; j < 8; j++) {
      const bf16* bp = F2Wt + (16 * j + m) * 256 + c * 32 + hi * 8;
      acc2[j] = wmma_bf16(a, ldb(bp), acc2[j]);
    }
  }

  // t2 = x + y ; BN2 partial sums ; write to out
  const int rowbase = tile * 16 + hi * 8;
#pragma unroll
  for (int j = 0; j < 8; j++) {
    const int f = 16 * j + m;
    const float f2b = F2b[f], sc = scale1[f], sh = shift1[f];
    float s = 0.f, q = 0.f;
#pragma unroll
    for (int r = 0; r < 8; r++) {
      const int nd = rowbase + r;
      const float x = sc * tmp[(size_t)nd * DD + f] + sh;
      const float t = x + acc2[j][r] + f2b;
      out[(size_t)nd * DD + f] = t;
      s += t; q += t * t;
    }
    s += __shfl_xor(s, 16); q += __shfl_xor(q, 16);
    if (hi == 0) { atomicAdd(&sAcc[f], s); atomicAdd(&sAcc[128 + f], q); }
  }
  __syncthreads();
  for (int i = threadIdx.x; i < 256; i += 256) atomicAdd(&colsum2[i], sAcc[i]);
}

// ---------------- final in-place BN2 on d_out ----------------
__global__ void k_final(float* __restrict__ out, const float* __restrict__ scale2,
                        const float* __restrict__ shift2, int total)
{
  int i = blockIdx.x * blockDim.x + threadIdx.x;
  if (i < total) {
    int f = i & (DD - 1);
    out[i] = scale2[f] * out[i] + shift2[f];
  }
}

extern "C" void kernel_launch(void* const* d_in, const int* in_sizes, int n_in,
                              void* d_out, int out_size, void* d_ws, size_t ws_size,
                              hipStream_t stream) {
  (void)in_sizes; (void)n_in; (void)out_size; (void)ws_size;
  const float* h     = (const float*)d_in[0];
  const int*   nbr   = (const int*)d_in[1];
  const float* W1    = (const float*)d_in[2];
  const float* b1    = (const float*)d_in[3];
  const float* W2    = (const float*)d_in[4];
  const float* b2    = (const float*)d_in[5];
  const float* OW    = (const float*)d_in[6];
  const float* Ob    = (const float*)d_in[7];
  const float* bn1g  = (const float*)d_in[8];
  const float* bn1b  = (const float*)d_in[9];
  const float* F1W   = (const float*)d_in[10];
  const float* F1b   = (const float*)d_in[11];
  const float* F2W   = (const float*)d_in[12];
  const float* F2b   = (const float*)d_in[13];
  const float* bn2g  = (const float*)d_in[14];
  const float* bn2b  = (const float*)d_in[15];
  float* out = (float*)d_out;

  // workspace carve (all 512B-multiple offsets)
  char* p = (char*)d_ws;
  bf16* W1t  = (bf16*)p; p += 128 * 128 * 2;   // 32 KB
  bf16* OWt  = (bf16*)p; p += 128 * 256 * 2;   // 64 KB (rows = out feature, cols = 256 concat)
  bf16* F1Wt = (bf16*)p; p += 256 * 128 * 2;   // 64 KB
  bf16* F2Wt = (bf16*)p; p += 128 * 256 * 2;   // 64 KB
  float* colsum1 = (float*)p; p += 256 * 4;
  float* colsum2 = (float*)p; p += 256 * 4;
  float* scale1  = (float*)p; p += 128 * 4;
  float* shift1  = (float*)p; p += 128 * 4;
  float* scale2  = (float*)p; p += 128 * 4;
  float* shift2  = (float*)p; p += 128 * 4;
  float* stats   = (float*)p; p += (size_t)NN * 3 * 4;
  float* tmp     = (float*)p; p += (size_t)NN * DD * 4;

  k_zero<<<1, 512, 0, stream>>>(colsum1, 512);  // colsum1+colsum2 contiguous
  k_transpose_bf16<<<(128 * 128 + 255) / 256, 256, 0, stream>>>(W1,  W1t,  128, 128);
  k_transpose_bf16<<<(256 * 128 + 255) / 256, 256, 0, stream>>>(OW,  OWt,  256, 128);
  k_transpose_bf16<<<(128 * 256 + 255) / 256, 256, 0, stream>>>(F1W, F1Wt, 128, 256);
  k_transpose_bf16<<<(256 * 128 + 255) / 256, 256, 0, stream>>>(F2W, F2Wt, 256, 128);

  k_edge<<<NN / 8, 256, 0, stream>>>(h, nbr, W1t, b1, W2, b2, stats);
  k_out<<<NN / 128, 256, 0, stream>>>(h, stats, OWt, Ob, tmp, colsum1);
  k_bnscale<<<1, 128, 0, stream>>>(colsum1, bn1g, bn1b, scale1, shift1);
  k_ffn<<<NN / 128, 256, 0, stream>>>(tmp, scale1, shift1, F1Wt, F1b, F2Wt, F2b, out, colsum2);
  k_bnscale<<<1, 128, 0, stream>>>(colsum2, bn2g, bn2b, scale2, shift2);
  k_final<<<(NN * DD + 255) / 256, 256, 0, stream>>>(out, scale2, shift2, NN * DD);
}
